// MultiHeadAttentionBlock_91070486544955
// MI455X (gfx1250) — compile-verified, matrix operand order checked by reading, not run on hardware
//
#include <hip/hip_runtime.h>
#include <math.h>

typedef __bf16 bf16;
typedef __attribute__((ext_vector_type(16))) __bf16 v16bf;
typedef __attribute__((ext_vector_type(4)))  __bf16 bf16x4;
typedef __attribute__((ext_vector_type(2)))  __bf16 bf16x2;
typedef __attribute__((ext_vector_type(8)))  float  v8f;
typedef int v4i_t __attribute__((vector_size(16)));

constexpr int NB  = 4;
constexpr int NS  = 2048;
constexpr int ND  = 1024;
constexpr int NH  = 16;
constexpr int NDK = 64;

// LDS row strides (elements), padded for bank-conflict-free fragment loads
constexpr int GLD = 34;   // GEMM tiles: 68B lane stride
constexpr int FLD = 72;   // flash tiles: 144B lane stride

// ---- CDNA5 async global->LDS copy (guarded; sync fallback) ----------------
#if defined(__gfx1250__) && __has_builtin(__builtin_amdgcn_global_load_async_to_lds_b128)
#define HAVE_ASYNC_COPY 1
#else
#define HAVE_ASYNC_COPY 0
#endif

#if defined(__gfx1250__) && __has_builtin(__builtin_amdgcn_s_wait_asynccnt)
#define WAIT_ASYNC(n) __builtin_amdgcn_s_wait_asynccnt(n)
#else
#define WAIT_ASYNC(n) asm volatile("s_wait_asynccnt %0" ::"i"(n) : "memory")
#endif

__device__ __forceinline__ void copy16(const bf16* g, bf16* l) {
#if HAVE_ASYNC_COPY
  __builtin_amdgcn_global_load_async_to_lds_b128(
      (__attribute__((address_space(1))) v4i_t*)(g),
      (__attribute__((address_space(3))) v4i_t*)(l),
      0, 0);
#else
  *(int4*)l = *(const int4*)g;
#endif
}

__device__ __forceinline__ void wait_async_le8() {
#if HAVE_ASYNC_COPY
  WAIT_ASYNC(8);
#endif
}
__device__ __forceinline__ void wait_async_le0() {
#if HAVE_ASYNC_COPY
  WAIT_ASYNC(0);
#endif
}

// ---- WMMA fragment helpers (ISA 7.12.2 layouts) ---------------------------
// lanes 0-15: K = {0..7,16..23}; lanes 16-31: K = {8..15,24..31}; 2 per VGPR.
__device__ __forceinline__ int frag_k(int lane, int v) {
  return ((v & 4) << 2) + ((lane >> 4) << 3) + ((v & 3) << 1);
}

// 16x32 fragment: per-lane row (A: M, B: N) with 32 contiguous K, stride ld.
__device__ __forceinline__ v16bf load_frag(const bf16* src, int ld) {
  const int lane = threadIdx.x & 31;
  const int row  = lane & 15;
  v16bf f;
#pragma unroll
  for (int v = 0; v < 8; ++v) {
    const int k = frag_k(lane, v);
    bf16x2 p = *(const bf16x2*)(src + row * ld + k);
    f[2 * v]     = p.x;
    f[2 * v + 1] = p.y;
  }
  return f;
}

__device__ __forceinline__ v8f wmma_bf16(v16bf a, v16bf b, v8f c) {
  return __builtin_amdgcn_wmma_f32_16x16x32_bf16(
      false, a, false, b, (short)0, c, false, false);
}

template <typename TIN>
__device__ __forceinline__ bf16x4 load4_bf16(const TIN* p) {
  if constexpr (__is_same(TIN, float)) {
    const float4 f = *(const float4*)p;
    bf16x4 o = {(bf16)f.x, (bf16)f.y, (bf16)f.z, (bf16)f.w};
    return o;
  } else {
    return *(const bf16x4*)p;
  }
}

// ---------------------------------------------------------------------------
// Tiled GEMM: Y(M=8192,N=1024) = X @ W + bias.  256 threads, 128x128 tile,
// 8 waves x (16 rows x 128 cols): 8 WMMAs per 32-deep K chunk per wave.
// OUT_MODE 0: bf16 (B,H,S,DK) | 1: bf16 (B,H,DK,S) | 2: f32 (M,N)
// ---------------------------------------------------------------------------
template <typename TIN, int OUT_MODE>
__global__ __launch_bounds__(256) void gemm128x128(
    const TIN* __restrict__ X, const float* __restrict__ W,
    const float* __restrict__ bias, void* __restrict__ Yv) {
  __shared__ __align__(16) bf16 lds_x[128 * GLD];   // [row][k]
  __shared__ __align__(16) bf16 lds_w[128 * GLD];   // transposed [n][k]

  const int tid  = threadIdx.x;
  const int lane = tid & 31;
  const int wave = tid >> 5;
  const int m0 = (blockIdx.x & 63) * 128;   // 64 m-blocks
  const int n0 = (blockIdx.x >> 6) * 128;   // 8 n-blocks

  const v8f zero8 = {0.f, 0.f, 0.f, 0.f, 0.f, 0.f, 0.f, 0.f};
  v8f acc[8] = {zero8, zero8, zero8, zero8, zero8, zero8, zero8, zero8};

  for (int kk = 0; kk < ND; kk += 32) {
    // X tile 128x32 -> bf16 LDS, vectorized 8B stores (16 elems/thread)
#pragma unroll
    for (int i = 0; i < 4; ++i) {
      const int idx = tid + i * 256;          // 0..1023
      const int row = idx >> 3;
      const int c4  = (idx & 7) << 2;
      *(bf16x4*)&lds_x[row * GLD + c4] =
          load4_bf16(X + (size_t)(m0 + row) * ND + kk + c4);
    }
    // W tile 32x128 -> transposed LDS [n][k] (16 elems/thread)
#pragma unroll
    for (int i = 0; i < 4; ++i) {
      const int idx = tid + i * 256;          // 0..1023
      const int k   = idx >> 5;               // 0..31
      const int c4  = (idx & 31) << 2;        // 0..124
      const float4 wf = *(const float4*)(W + (size_t)(kk + k) * ND + n0 + c4);
      lds_w[(c4 + 0) * GLD + k] = (bf16)wf.x;
      lds_w[(c4 + 1) * GLD + k] = (bf16)wf.y;
      lds_w[(c4 + 2) * GLD + k] = (bf16)wf.z;
      lds_w[(c4 + 3) * GLD + k] = (bf16)wf.w;
    }
    if (kk + 32 < ND)
      __builtin_prefetch(X + (size_t)(m0 + (tid >> 3)) * ND + kk + 32, 0, 0);
    __syncthreads();

    const v16bf a = load_frag(lds_x + wave * 16 * GLD, GLD);
#pragma unroll
    for (int t = 0; t < 8; ++t) {
      const v16bf bfr = load_frag(lds_w + t * 16 * GLD, GLD);
      acc[t] = wmma_bf16(a, bfr, acc[t]);
    }
    __syncthreads();
  }

  const int lhalf = lane >> 4;
  const int lcol  = lane & 15;
#pragma unroll
  for (int t = 0; t < 8; ++t) {
    const int n = n0 + t * 16 + lcol;
    const float bv = bias[n];
#pragma unroll
    for (int r = 0; r < 8; ++r) {
      const int gm = m0 + wave * 16 + r + lhalf * 8;
      const float val = acc[t][r] + bv;
      if (OUT_MODE == 2) {
        ((float*)Yv)[(size_t)gm * ND + n] = val;
      } else {
        const int b  = gm / NS;
        const int s  = gm % NS;
        const int h  = n >> 6;
        const int dk = n & 63;
        bf16* Y = (bf16*)Yv;
        if (OUT_MODE == 0)
          Y[(((size_t)(b * NH + h)) * NS + s) * NDK + dk] = (bf16)val;
        else
          Y[(((size_t)(b * NH + h)) * NDK + dk) * NS + s] = (bf16)val;
      }
    }
  }
}

// ---------------------------------------------------------------------------
// Flash attention: 128 threads = 4 waves handle 64 queries of one (b,h).
// K/V tiles (64 keys) double-buffered in LDS via async global->LDS copies.
// Q,K: (B,H,S,DK) bf16.  Vt: (B,H,DK,S) bf16.  ctx out: (B,S,D) bf16.
// ---------------------------------------------------------------------------
__global__ __launch_bounds__(128) void flash_kernel(
    const bf16* __restrict__ Q, const bf16* __restrict__ K,
    const bf16* __restrict__ Vt, const int* __restrict__ mask,
    bf16* __restrict__ ctx) {
  __shared__ __align__(16) bf16 lds_k[2][64 * FLD];  // [key][dk]
  __shared__ __align__(16) bf16 lds_v[2][64 * FLD];  // [dk][key]
  __shared__ __align__(16) bf16 lds_p[4][16 * FLD];  // per-wave P tile

  const int tid  = threadIdx.x;
  const int lane = tid & 31;
  const int wave = tid >> 5;
  const int qb = blockIdx.x & 31;     // S/64 query blocks
  const int bh = blockIdx.x >> 5;     // 0..63
  const int b = bh >> 4;
  const int h = bh & 15;
  const int q0 = qb * 64 + wave * 16;

  const bf16* Qb = Q  + (size_t)bh * NS * NDK + (size_t)q0 * NDK;
  const bf16* Kb = K  + (size_t)bh * NS * NDK;
  const bf16* Vb = Vt + (size_t)bh * NDK * NS;
  const int* maskb = mask + b * NS;

  const v16bf aq0 = load_frag(Qb, NDK);        // dk 0..31
  const v16bf aq1 = load_frag(Qb + 32, NDK);   // dk 32..63

  const v8f zero8 = {0.f, 0.f, 0.f, 0.f, 0.f, 0.f, 0.f, 0.f};
  v8f acc[4] = {zero8, zero8, zero8, zero8};
  float mrow[8], lrow[8];
#pragma unroll
  for (int r = 0; r < 8; ++r) { mrow[r] = -1e30f; lrow[r] = 0.f; }

  const int lhalf = lane >> 4;
  const int lcol  = lane & 15;
  const float scale = 0.125f;   // 1/sqrt(DK)

  // Cooperative stage of one 64-key chunk: 8 x 16B per thread (K + V^T rows)
  auto stage = [&](int buf, int kt) {
#pragma unroll
    for (int j = 0; j < 4; ++j) {
      const int o = tid + j * 128;            // 0..511
      const int row = o >> 3, c = o & 7;      // 64 rows x 8 x 16B
      copy16(Kb + (size_t)(kt + row) * NDK + c * 8,
             &lds_k[buf][row * FLD + c * 8]);
    }
#pragma unroll
    for (int j = 0; j < 4; ++j) {
      const int o = tid + j * 128;
      const int row = o >> 3, c = o & 7;      // row = dk
      copy16(Vb + (size_t)row * NS + kt + c * 8,
             &lds_v[buf][row * FLD + c * 8]);
    }
  };

  int buf = 0;
  stage(0, 0);
  for (int kt = 0; kt < NS; kt += 64) {
    if (kt + 64 < NS) {
      stage(buf ^ 1, kt + 64);
      wait_async_le8();      // previous chunk's 8 async ops complete
    } else {
      wait_async_le0();
    }
    __syncthreads();

    const bf16* kb = lds_k[buf];
    const bf16* vb = lds_v[buf];

    // scores for 4 x 16-key subtiles (Q 16x64 @ K^T)
    v8f s[4];
    int mv[4];
#pragma unroll
    for (int sub = 0; sub < 4; ++sub) {
      const v16bf bk0 = load_frag(kb + (sub * 16) * FLD, FLD);
      const v16bf bk1 = load_frag(kb + (sub * 16) * FLD + 32, FLD);
      v8f t = zero8;
      t = wmma_bf16(aq0, bk0, t);
      t = wmma_bf16(aq1, bk1, t);
      s[sub] = t;
      mv[sub] = maskb[kt + sub * 16 + lcol];
    }

    // online softmax over all 64 keys (one reduction pass per row)
    bf16* pw = lds_p[wave];
#pragma unroll
    for (int r = 0; r < 8; ++r) {
      float x[4];
      float xm = -1e30f;
#pragma unroll
      for (int sub = 0; sub < 4; ++sub) {
        x[sub] = (mv[sub] == 0) ? -1e30f : s[sub][r] * scale;
        xm = fmaxf(xm, x[sub]);
      }
      xm = fmaxf(xm, __shfl_xor(xm, 1, 16));
      xm = fmaxf(xm, __shfl_xor(xm, 2, 16));
      xm = fmaxf(xm, __shfl_xor(xm, 4, 16));
      xm = fmaxf(xm, __shfl_xor(xm, 8, 16));
      const float mnew = fmaxf(mrow[r], xm);
      const float corr = __expf(mrow[r] - mnew);
      float psum = 0.f;
#pragma unroll
      for (int sub = 0; sub < 4; ++sub) {
        const float p = __expf(x[sub] - mnew);
        psum += p;
        pw[(r + lhalf * 8) * FLD + sub * 16 + lcol] = (bf16)p;
      }
      psum += __shfl_xor(psum, 1, 16);
      psum += __shfl_xor(psum, 2, 16);
      psum += __shfl_xor(psum, 4, 16);
      psum += __shfl_xor(psum, 8, 16);
      lrow[r] = lrow[r] * corr + psum;
      mrow[r] = mnew;
#pragma unroll
      for (int t = 0; t < 4; ++t) acc[t][r] *= corr;
    }

    // P(16x64) @ V(64x64): LDS ops are in-order within a wave, and pw is
    // private to this wave, so no barrier needed before re-reading it.
    const v16bf ap0 = load_frag(pw, FLD);
    const v16bf ap1 = load_frag(pw + 32, FLD);
#pragma unroll
    for (int t = 0; t < 4; ++t) {
      acc[t] = wmma_bf16(ap0, load_frag(vb + (t * 16) * FLD, FLD), acc[t]);
      acc[t] = wmma_bf16(ap1, load_frag(vb + (t * 16) * FLD + 32, FLD), acc[t]);
    }
    __syncthreads();   // all waves done with kv buffers before restaging
    buf ^= 1;
  }

  // normalize + merge heads -> ctx (B,S,D)
#pragma unroll
  for (int r = 0; r < 8; ++r) {
    const float inv = 1.0f / lrow[r];
    const int sq = q0 + r + lhalf * 8;
#pragma unroll
    for (int t = 0; t < 4; ++t) {
      const int col = h * 64 + t * 16 + lcol;
      ctx[((size_t)b * NS + sq) * ND + col] = (bf16)(acc[t][r] * inv);
    }
  }
}

// ---------------------------------------------------------------------------
extern "C" void kernel_launch(void* const* d_in, const int* in_sizes, int n_in,
                              void* d_out, int out_size, void* d_ws,
                              size_t ws_size, hipStream_t stream) {
  (void)in_sizes; (void)n_in; (void)out_size; (void)ws_size;
  const float* q  = (const float*)d_in[0];
  const float* k  = (const float*)d_in[1];
  const float* v  = (const float*)d_in[2];
  const int*  msk = (const int*)d_in[3];
  const float* Wq = (const float*)d_in[4];
  const float* bq = (const float*)d_in[5];
  const float* Wk = (const float*)d_in[6];
  const float* bk = (const float*)d_in[7];
  const float* Wv = (const float*)d_in[8];
  const float* bv = (const float*)d_in[9];
  const float* Wo = (const float*)d_in[10];
  const float* bo = (const float*)d_in[11];
  float* out = (float*)d_out;

  const size_t seg = (size_t)NB * NH * NS * NDK * sizeof(bf16);  // 16 MB
  bf16* Qbf = (bf16*)d_ws;
  bf16* Kbf = (bf16*)((char*)d_ws + seg);
  bf16* Vt  = (bf16*)((char*)d_ws + 2 * seg);
  bf16* ctx = (bf16*)((char*)d_ws + 3 * seg);

  const dim3 gg(512), gb(256);
  gemm128x128<float, 0><<<gg, gb, 0, stream>>>(q, Wq, bq, (void*)Qbf);
  gemm128x128<float, 0><<<gg, gb, 0, stream>>>(k, Wk, bk, (void*)Kbf);
  gemm128x128<float, 1><<<gg, gb, 0, stream>>>(v, Wv, bv, (void*)Vt);
  flash_kernel<<<dim3(NB * NH * (NS / 64)), dim3(128), 0, stream>>>(
      Qbf, Kbf, Vt, msk, ctx);
  gemm128x128<bf16, 2><<<gg, gb, 0, stream>>>(ctx, Wo, bo, (void*)out);
}
